// HermiteConv2dKan_68521908240750
// MI455X (gfx1250) — matrix operand order repaired and checked by reading, not compiled, not run on hardware
//
#include <hip/hip_runtime.h>
#include <hip/hip_bf16.h>
#include <math.h>

// Hermite-KAN conv as a fused f16-WMMA GEMM.
//   M = 32 out-ch, N = 4096 (4 batches * 32*32 pixels), K = 720 -> reordered/padded to 800.
// K-axis ordering (ours to choose; A and B agree):
//   m = np*160 + ik,  np in {0:mish, 1:H0, 2:H1, 3:H2, 4:H3},  ik = i*9 + k (0..143, padded to 160)
// 160 = 5*32, so feature group np occupies K-chunks [np*5, np*5+5) exactly, and the
// (lane, j) fragment slot of a given ik is identical across all 5 groups.

typedef _Float16 half16 __attribute__((ext_vector_type(16)));
typedef float    float8 __attribute__((ext_vector_type(8)));

#define KCHUNKS 25      // 800 / 32
#define IKPAD   160
#define IKVAL   144

__global__ __launch_bounds__(128)
void hermite_kan_wmma_kernel(const float* __restrict__ x,   // (4,16,32,32)
                             const float* __restrict__ w,   // (16,32,9)
                             const float* __restrict__ c,   // (16,32,9,4)
                             float* __restrict__ out)       // (4,32,32,32)
{
    // A fragments (fused weights): [kb][mt][lane][j], f16.  25*2*32 half16 = 51200 B
    __shared__ half16 sA[KCHUNKS * 2 * 32];
    // B fragments (features):      [kb][wt][lane][j], f16.  25*4*32 half16 = 102400 B
    __shared__ half16 sB[KCHUNKS * 4 * 32];
    // Zero-padded x tile: 16 ch x 4 rows x 34 cols f32 = 8704 B
    __shared__ float sx[16][4][34];

    const int tid  = threadIdx.x;
    const int b    = blockIdx.x >> 4;        // batch
    const int tile = blockIdx.x & 15;        // 16 tiles of 64 columns per batch
    const int l0   = tile * 64;              // first flat pixel
    const int y0   = tile * 2;               // first image row (64 cols = 2 rows)

    // ---- Phase 1: stage zero-padded input tile (rows y0-1..y0+2, cols -1..32) ----
    for (int e = tid; e < 16 * 4 * 34; e += 128) {
        int cx  = e % 34;
        int t   = e / 34;
        int ry  = t & 3;
        int ich = t >> 2;
        int row = y0 + ry - 1;
        int col = cx - 1;
        float v = 0.0f;
        if (row >= 0 && row < 32 && col >= 0 && col < 32)
            v = x[((b * 16 + ich) * 32 + row) * 32 + col];
        sx[ich][ry][cx] = v;
    }

    // ---- Phase 2: fused weights -> A-fragment layout, branch-free ----
    // One (o, ik) pair per iteration; 5 stores at stride 5120 halves (one per group).
    // 16-bit A 16x32 lane map: K = j + 8*(j>=8) + 8*(lane>=16)
    //   inverse: lamhi = (K>>3)&1 ; j = (K&7) + ((K>>4)<<3)
    {
        _Float16* sAh = (_Float16*)sA;
        for (int e = tid; e < 32 * IKPAD; e += 128) {
            int o   = e & 31;
            int ik  = e >> 5;                 // 0..159
            int ikc = (ik < IKVAL) ? ik : (IKVAL - 1);
            int i   = ikc / 9;
            int k   = ikc - 9 * i;
            int wi  = (i * 32 + o) * 9 + k;
            float wv = (ik < IKVAL) ? w[wi] : 0.0f;
            const float* cp = c + wi * 4;
            float a0 = wv;                    // mish group
            float a1 = wv * cp[0];            // H0
            float a2 = wv * cp[1];            // H1
            float a3 = wv * cp[2];            // H2
            float a4 = wv * cp[3];            // H3

            int mt    = o >> 4;
            int Kl    = ik & 31;
            int lamhi = (Kl >> 3) & 1;
            int j     = (Kl & 7) + ((Kl >> 4) << 3);
            int lam   = (o & 15) + (lamhi << 4);
            int kb0   = ik >> 5;
            int addr  = ((kb0 * 2 + mt) * 32 + lam) * 16 + j;
            sAh[addr            ] = (_Float16)a0;
            sAh[addr +  5 * 1024] = (_Float16)a1;   // 5*(2*32*16) = 5120
            sAh[addr + 10 * 1024] = (_Float16)a2;
            sAh[addr + 15 * 1024] = (_Float16)a3;
            sAh[addr + 20 * 1024] = (_Float16)a4;
        }
    }
    __syncthreads();

    // ---- Phase 3: features -> B-fragment layout, branch-free, one mish per patch value ----
    // 16-bit B 32x16 lane map: K = j + 8*(j>=8) + 8*(lane>=16) (same as A).
    {
        _Float16* sBh = (_Float16*)sB;
        for (int e = tid; e < 64 * IKPAD; e += 128) {
            int col = e & 63;
            int ik  = e >> 6;                 // 0..159
            int ikc = (ik < IKVAL) ? ik : (IKVAL - 1);   // pad slots: finite dummy (A is 0 there)
            int i   = ikc / 9;
            int k   = ikc - 9 * i;
            int ky  = k / 3;
            int kx  = k - 3 * ky;
            int yr  = col >> 5;
            int xc  = col & 31;
            float p = sx[i][yr + ky][xc + kx];

            float sp = log1pf(expf(p));
            float f0 = p * tanhf(sp);                 // mish
            float p2 = p * p;
            float f2 = 2.0f * p;                      // H1
            float f3 = 4.0f * p2 - 2.0f;              // H2
            float f4 = p * (8.0f * p2 - 12.0f);       // H3

            int wt    = col >> 4;
            int lamhi = (ik >> 3) & 1;
            int j     = (ik & 7) + (((ik >> 4) & 1) << 3);
            int lam   = (col & 15) + (lamhi << 4);
            int kb0   = ik >> 5;
            int addr  = ((kb0 * 4 + wt) * 32 + lam) * 16 + j;
            sBh[addr            ] = (_Float16)f0;
            sBh[addr + 10 * 1024] = (_Float16)1.0f;   // H0; 5*(4*32*16) = 10240
            sBh[addr + 20 * 1024] = (_Float16)f2;
            sBh[addr + 30 * 1024] = (_Float16)f3;
            sBh[addr + 40 * 1024] = (_Float16)f4;
        }
    }
    __syncthreads();

    // ---- Phase 4: pure ds_load + WMMA hot loop ----
    const int lane = tid & 31;
    const int wt   = tid >> 5;               // wave id 0..3 -> N-tile

    float8 acc0 = {};                        // out channels 0..15
    float8 acc1 = {};                        // out channels 16..31

    for (int kb = 0; kb < KCHUNKS; ++kb) {
        half16 a0 = sA[(kb * 2 + 0) * 32 + lane];
        half16 a1 = sA[(kb * 2 + 1) * 32 + lane];
        half16 bf = sB[(kb * 4 + wt) * 32 + lane];
        acc0 = __builtin_amdgcn_wmma_f32_16x16x32_f16(
            false, a0, false, bf, (short)0, acc0, false, false);
        acc1 = __builtin_amdgcn_wmma_f32_16x16x32_f16(
            false, a1, false, bf, (short)0, acc1, false, false);
    }

    // ---- Phase 5: store D (f32 16x16 C/D layout: M = r + 8*(lane>=16), N = lane%16) ----
    const int ct   = wt * 16 + (lane & 15);
    const int l    = l0 + ct;
    const int base = b * 32 * 1024;
    const int oofs = (lane >= 16) ? 8 : 0;
    #pragma unroll
    for (int r = 0; r < 8; ++r) {
        int o = r + oofs;
        out[base + o * 1024 + l]        = acc0[r];
        out[base + (o + 16) * 1024 + l] = acc1[r];
    }
}

extern "C" void kernel_launch(void* const* d_in, const int* in_sizes, int n_in,
                              void* d_out, int out_size, void* d_ws, size_t ws_size,
                              hipStream_t stream) {
    const float* x = (const float*)d_in[0];   // (4,16,32,32)
    const float* w = (const float*)d_in[1];   // (16,32,9)
    const float* c = (const float*)d_in[2];   // (16,32,9,4)
    float* out = (float*)d_out;               // (4,32,32,32)
    (void)in_sizes; (void)n_in; (void)out_size; (void)d_ws; (void)ws_size;

    // 64 workgroups x 128 threads (4 waves); each WG does 64 of 4096 columns.
    hermite_kan_wmma_kernel<<<64, 128, 0, stream>>>(x, w, c, out);
}
